// StabilityPredictorPooling_13443247636503
// MI455X (gfx1250) — matrix operand order repaired and checked by reading, not compiled, not run on hardware
//
#include <hip/hip_runtime.h>
#include <math.h>

typedef _Float16 v16h __attribute__((ext_vector_type(16)));
typedef float    v8f  __attribute__((ext_vector_type(8)));

constexpr int Bc = 2, Lc = 1024, Kc = 32, Hc = 128;
constexpr int NNODE = Bc * Lc;          // 2048
constexpr int NEDGE = NNODE * Kc;       // 65536

__constant__ int PAIR_A[25] = {0,1,2,3,4,0,0,0,0,1,1,1,4,4,3,1,2,3,4,2,3,4,2,3,2};
__constant__ int PAIR_B[25] = {0,1,2,3,4,1,2,3,4,2,3,4,2,3,2,0,0,0,0,1,1,1,4,4,3};

__device__ __forceinline__ float gelu_f(float x) {
    return 0.5f * x * (1.0f + erff(x * 0.7071067811865476f));
}

__device__ __forceinline__ float wave_sum(float v) {
#pragma unroll
    for (int o = 16; o > 0; o >>= 1) v += __shfl_xor(v, o, 32);
    return v;
}

// LayerNorm over 128 cols, lane handles cols lane, lane+32, lane+64, lane+96
__device__ __forceinline__ void ln4(const float v[4], int lane, const float* g, const float* bt, float o[4]) {
    float s = v[0] + v[1] + v[2] + v[3];
    s = wave_sum(s);
    float mean = s * (1.0f / 128.0f);
    float q = 0.f;
#pragma unroll
    for (int t = 0; t < 4; ++t) { float d = v[t] - mean; q += d * d; }
    q = wave_sum(q);
    float inv = rsqrtf(q * (1.0f / 128.0f) + 1e-5f);
#pragma unroll
    for (int t = 0; t < 4; ++t) { int c = lane + 32 * t; o[t] = (v[t] - mean) * inv * g[c] + bt[c]; }
}

// ---- WMMA fragment loads (per CDNA5 ISA 7.12.2 layouts) ----
__device__ __forceinline__ v16h wmma_load_a(const _Float16* A, int lda, int m0, int k0, int lane) {
    const _Float16* p = A + (m0 + (lane & 15)) * lda + k0 + ((lane >> 4) << 3);
    v16h a;
#pragma unroll
    for (int j = 0; j < 8; ++j) { a[j] = p[j]; a[j + 8] = p[j + 16]; }
    return a;
}
__device__ __forceinline__ v16h wmma_load_b(const _Float16* W, int N, int k0, int n0, int lane) {
    const _Float16* p = W + (k0 + ((lane >> 4) << 3)) * N + n0 + (lane & 15);
    v16h b;
#pragma unroll
    for (int j = 0; j < 8; ++j) { b[j] = p[j * N]; b[j + 8] = p[(j + 16) * N]; }
    return b;
}

// Block-wide GEMM: OUT[(MT*16) x N] = act(A @ W + bias)
// A in LDS f16 (row stride lda); W f16 global row-major [K][N].
// 8 waves; wave w owns column slabs n0 = w*16 + 128*t. K mult of 32, N mult of 128.
// B (weight) fragment is loaded ONCE per k-step and reused across the MT row-tiles,
// so two WMMAs issue back-to-back per global fragment in the M=32 kernels.
template<bool GELU, bool O16, bool O32, int MT>
__device__ void block_gemm(const _Float16* A, int lda,
                           const _Float16* W, int K, int N, const float* bias,
                           _Float16* o16, float* o32, int ldo) {
    int lane = threadIdx.x & 31, wv = threadIdx.x >> 5;
    for (int n0 = wv * 16; n0 < N; n0 += 128) {
        v8f acc[MT];
#pragma unroll
        for (int mt = 0; mt < MT; ++mt) acc[mt] = (v8f){};
        for (int k0 = 0; k0 < K; k0 += 32) {
            v16h b = wmma_load_b(W, N, k0, n0, lane);
#pragma unroll
            for (int mt = 0; mt < MT; ++mt) {
                v16h a = wmma_load_a(A, lda, mt * 16, k0, lane);
                acc[mt] = __builtin_amdgcn_wmma_f32_16x16x32_f16(false, a, false, b, (short)0, acc[mt], false, false);
            }
        }
        int n = n0 + (lane & 15);
        float bv = bias ? bias[n] : 0.f;
#pragma unroll
        for (int mt = 0; mt < MT; ++mt) {
            int mb = mt * 16 + ((lane >> 4) << 3);
#pragma unroll
            for (int r = 0; r < 8; ++r) {
                float x = acc[mt][r] + bv;
                if (GELU) x = gelu_f(x);
                if (O32) o32[(mb + r) * ldo + n] = x;
                if (O16) o16[(mb + r) * ldo + n] = (_Float16)x;
            }
        }
    }
}

// ---------------- misc small kernels ----------------
__global__ void k_cvt(const float* __restrict__ s, _Float16* __restrict__ d, int n) {
    int i = blockIdx.x * blockDim.x + threadIdx.x;
    if (i < n) d[i] = (_Float16)s[i];
}

__global__ void k_backbone(const float* __restrict__ X, float* __restrict__ Cb) {
    int i = blockIdx.x * blockDim.x + threadIdx.x;
    if (i >= NNODE) return;
    const float* x = X + (size_t)i * 12;
    float Nx = x[0], Ny = x[1], Nz = x[2];
    float Cax = x[3], Cay = x[4], Caz = x[5];
    float Cx = x[6], Cy = x[7], Cz = x[8];
    float bx = Cax - Nx, by = Cay - Ny, bz = Caz - Nz;
    float cx = Cx - Cax, cy = Cy - Cay, cz = Cz - Caz;
    float ax = by * cz - bz * cy, ay = bz * cx - bx * cz, az = bx * cy - by * cx;
    Cb[i * 3 + 0] = -0.58273431f * ax + 0.56802827f * bx - 0.54067466f * cx + Cax;
    Cb[i * 3 + 1] = -0.58273431f * ay + 0.56802827f * by - 0.54067466f * cy + Cay;
    Cb[i * 3 + 2] = -0.58273431f * az + 0.56802827f * bz - 0.54067466f * cz + Caz;
}

__global__ void k_topk(const float* __restrict__ X, const float* __restrict__ mask, int* __restrict__ Eidx) {
    __shared__ float Dv[Lc];
    __shared__ float rv[256];
    __shared__ int   ri[256];
    int bi = blockIdx.x, b = bi >> 10, tid = threadIdx.x;
    const float* ci = X + ((size_t)bi * 4 + 1) * 3;
    float cx = ci[0], cy = ci[1], cz = ci[2];
    float mi = mask[bi];
    float lmax = -3.4e38f;
    for (int j = tid; j < Lc; j += 256) {
        const float* cj = X + (((size_t)(b * Lc + j)) * 4 + 1) * 3;
        float dx = cx - cj[0], dy = cy - cj[1], dz = cz - cj[2];
        float d = sqrtf(dx * dx + dy * dy + dz * dz + 1e-6f);
        Dv[j] = d;
        lmax = fmaxf(lmax, d);
    }
    rv[tid] = lmax;
    __syncthreads();
    for (int s = 128; s > 0; s >>= 1) {
        if (tid < s) rv[tid] = fmaxf(rv[tid], rv[tid + s]);
        __syncthreads();
    }
    float rmax = rv[0];
    __syncthreads();
    for (int j = tid; j < Lc; j += 256) {
        float m2 = mi * mask[b * Lc + j];
        Dv[j] = Dv[j] * m2 + (1.f - m2) * rmax;
    }
    __syncthreads();
    for (int k = 0; k < Kc; ++k) {
        float best = 3.4e38f; int bidx = Lc;
        for (int j = tid; j < Lc; j += 256) {
            float v = Dv[j];
            if (v < best) { best = v; bidx = j; }
        }
        rv[tid] = best; ri[tid] = bidx;
        __syncthreads();
        for (int s = 128; s > 0; s >>= 1) {
            if (tid < s) {
                float vo = rv[tid + s]; int io = ri[tid + s];
                if (vo < rv[tid] || (vo == rv[tid] && io < ri[tid])) { rv[tid] = vo; ri[tid] = io; }
            }
            __syncthreads();
        }
        if (tid == 0) { Eidx[bi * Kc + k] = ri[0]; Dv[ri[0]] = 3.4e38f; }
        __syncthreads();
    }
}

__global__ void k_rank(const float* __restrict__ chainM, const float* __restrict__ mask, int* __restrict__ rnk) {
    int i = blockIdx.x * blockDim.x + threadIdx.x;
    if (i >= NNODE) return;
    int b = i >> 10, l = i & 1023;
    float vi = (chainM[i] * mask[i] + 1e-4f) * (float)(l + 1);
    const float* cm = chainM + b * Lc;
    const float* mk = mask + b * Lc;
    int r = 0;
    for (int j = 0; j < Lc; ++j) {
        float vj = (cm[j] * mk[j] + 1e-4f) * (float)(j + 1);
        if (vj < vi || (vj == vi && j < l)) ++r;
    }
    rnk[i] = r;
}

__device__ __forceinline__ float3 get_atom(const float* X, const float* Cb, int g, int a) {
    if (a == 4) { const float* p = Cb + (size_t)g * 3; return make_float3(p[0], p[1], p[2]); }
    int slot = (a == 0) ? 1 : (a == 1) ? 0 : a;   // atoms = [Ca,N,C,O,Cb] -> X slots [1,0,2,3]
    const float* p = X + ((size_t)g * 4 + slot) * 3;
    return make_float3(p[0], p[1], p[2]);
}

// ---------------- edge features + projections ----------------
__global__ void k_edgefeat(const float* __restrict__ X, const float* __restrict__ Cb,
                           const int* __restrict__ Eidx, const int* __restrict__ ridx,
                           const int* __restrict__ cenc, const float* __restrict__ posW,
                           const float* __restrict__ posB, const _Float16* __restrict__ hWef,
                           const float* __restrict__ neg, const float* __restrict__ neb,
                           const _Float16* __restrict__ hWe, const float* __restrict__ Web,
                           float* __restrict__ hE) {
    __shared__ _Float16 feat[32 * 416];
    __shared__ float    Dp[32 * 25];
    __shared__ float    Et[32 * 128];
    __shared__ _Float16 A1[32 * 128];
    int bi = blockIdx.x, b = bi >> 10, tid = threadIdx.x;
    for (int idx = tid; idx < 32 * 25; idx += 256) {
        int e = idx / 25, p = idx % 25;
        int gj = b * Lc + Eidx[bi * Kc + e];
        float3 ai = get_atom(X, Cb, bi, PAIR_A[p]);
        float3 aj = get_atom(X, Cb, gj, PAIR_B[p]);
        float dx = ai.x - aj.x, dy = ai.y - aj.y, dz = ai.z - aj.z;
        Dp[idx] = sqrtf(dx * dx + dy * dy + dz * dz + 1e-6f);
    }
    __syncthreads();
    for (int idx = tid; idx < 32 * 416; idx += 256) {
        int e = idx / 416, c = idx % 416;
        float v;
        if (c < 16) {
            int gj = b * Lc + Eidx[bi * Kc + e];
            int d;
            if (cenc[bi] == cenc[gj]) {
                d = ridx[bi] - ridx[gj] + 32;
                d = d < 0 ? 0 : (d > 64 ? 64 : d);
            } else d = 65;
            v = posW[d * 16 + c] + posB[c];
        } else {
            int p = (c - 16) >> 4, r = (c - 16) & 15;
            float mu = 2.f + (20.f / 15.f) * (float)r;
            float t = (Dp[e * 25 + p] - mu) * 0.8f;   // sigma = 1.25
            v = expf(-t * t);
        }
        feat[e * 416 + c] = (_Float16)v;
    }
    __syncthreads();
    block_gemm<false, false, true, 2>(feat, 416, hWef, 416, 128, nullptr, nullptr, Et, 128);
    __syncthreads();
    int lane = tid & 31, wv = tid >> 5;
    for (int r4 = 0; r4 < 4; ++r4) {
        int e = wv * 4 + r4;
        float v[4], o[4];
#pragma unroll
        for (int t = 0; t < 4; ++t) v[t] = Et[e * 128 + lane + 32 * t];
        ln4(v, lane, neg, neb, o);
#pragma unroll
        for (int t = 0; t < 4; ++t) {
            int c = lane + 32 * t;
            Et[e * 128 + c] = o[t];
            A1[e * 128 + c] = (_Float16)o[t];
        }
    }
    __syncthreads();
    block_gemm<false, false, true, 2>(A1, 128, hWe, 128, 128, Web, nullptr, Et, 128);
    __syncthreads();
    for (int idx = tid; idx < 32 * 128; idx += 256)
        hE[(size_t)bi * 32 * 128 + idx] = Et[idx];
}

// ---------------- encoder message / edge update ----------------
__global__ void k_enc_msg(const float* __restrict__ hV, const float* __restrict__ hE,
                          const int* __restrict__ Eidx, const float* __restrict__ mask,
                          const _Float16* W1, const float* b1, const _Float16* W2, const float* b2,
                          const _Float16* W3, const float* b3, float* __restrict__ msum) {
    __shared__ _Float16 A0[32 * 384];
    __shared__ _Float16 A1[32 * 128];
    __shared__ float    Cm[32 * 128];
    __shared__ float    wAtt[32];
    int bi = blockIdx.x, b = bi >> 10, tid = threadIdx.x;
    float mi = mask[bi];
    if (tid < 32) wAtt[tid] = mi * mask[b * Lc + Eidx[bi * Kc + tid]];
    for (int idx = tid; idx < 32 * 384; idx += 256) {
        int e = idx / 384, c = idx % 384;
        float v;
        if (c < 128) v = hV[(size_t)bi * 128 + c];
        else if (c < 256) v = hE[((size_t)bi * 32 + e) * 128 + (c - 128)];
        else v = hV[(size_t)(b * Lc + Eidx[bi * Kc + e]) * 128 + (c - 256)];
        A0[idx] = (_Float16)v;
    }
    __syncthreads();
    block_gemm<true, true, false, 2>(A0, 384, W1, 384, 128, b1, A1, nullptr, 128);
    __syncthreads();
    block_gemm<true, true, false, 2>(A1, 128, W2, 128, 128, b2, A0, nullptr, 128);
    __syncthreads();
    block_gemm<false, false, true, 2>(A0, 128, W3, 128, 128, b3, nullptr, Cm, 128);
    __syncthreads();
    if (tid < 128) {
        float s = 0.f;
        for (int e = 0; e < 32; ++e) s += Cm[e * 128 + tid] * wAtt[e];
        msum[(size_t)bi * 128 + tid] = s;
    }
}

__global__ void k_enc_edge(const float* __restrict__ hV, float* __restrict__ hE,
                           const int* __restrict__ Eidx,
                           const _Float16* W1, const float* b1, const _Float16* W2, const float* b2,
                           const _Float16* W3, const float* b3,
                           const float* __restrict__ n3g, const float* __restrict__ n3b) {
    __shared__ _Float16 A0[32 * 384];
    __shared__ _Float16 A1[32 * 128];
    __shared__ float    Cm[32 * 128];
    int bi = blockIdx.x, b = bi >> 10, tid = threadIdx.x;
    for (int idx = tid; idx < 32 * 384; idx += 256) {
        int e = idx / 384, c = idx % 384;
        float v;
        if (c < 128) v = hV[(size_t)bi * 128 + c];
        else if (c < 256) v = hE[((size_t)bi * 32 + e) * 128 + (c - 128)];
        else v = hV[(size_t)(b * Lc + Eidx[bi * Kc + e]) * 128 + (c - 256)];
        A0[idx] = (_Float16)v;
    }
    __syncthreads();
    block_gemm<true, true, false, 2>(A0, 384, W1, 384, 128, b1, A1, nullptr, 128);
    __syncthreads();
    block_gemm<true, true, false, 2>(A1, 128, W2, 128, 128, b2, A0, nullptr, 128);
    __syncthreads();
    block_gemm<false, false, true, 2>(A0, 128, W3, 128, 128, b3, nullptr, Cm, 128);
    __syncthreads();
    int lane = tid & 31, wv = tid >> 5;
    for (int r4 = 0; r4 < 4; ++r4) {
        int e = wv * 4 + r4;
        float* row = hE + ((size_t)bi * 32 + e) * 128;
        float v[4], o[4];
#pragma unroll
        for (int t = 0; t < 4; ++t) { int c = lane + 32 * t; v[t] = row[c] + Cm[e * 128 + c]; }
        ln4(v, lane, n3g, n3b, o);
#pragma unroll
        for (int t = 0; t < 4; ++t) row[lane + 32 * t] = o[t];
    }
}

// ---------------- decoder message ----------------
__global__ void k_dec_msg(const float* __restrict__ hV, const float* __restrict__ hVenc,
                          const float* __restrict__ hE, const int* __restrict__ Eidx,
                          const int* __restrict__ rnk, const int* __restrict__ aa,
                          const float* __restrict__ mask, const float* __restrict__ Ws,
                          const _Float16* W1, const float* b1, const _Float16* W2, const float* b2,
                          const _Float16* W3, const float* b3, float* __restrict__ msum) {
    __shared__ _Float16 A0[32 * 512];
    __shared__ _Float16 A1[32 * 128];
    __shared__ float    Cm[32 * 128];
    int bi = blockIdx.x, b = bi >> 10, tid = threadIdx.x;
    float mi = mask[bi];
    int ri = rnk[bi];
    for (int idx = tid; idx < 32 * 512; idx += 256) {
        int e = idx >> 9, c = idx & 511;
        int gj = b * Lc + Eidx[bi * Kc + e];
        float att = (rnk[gj] < ri) ? 1.f : 0.f;
        float mbw = mi * att, mfw = mi * (1.f - att);
        float v;
        if (c < 128) v = hV[(size_t)bi * 128 + c];
        else if (c < 256) { float he = hE[((size_t)bi * 32 + e) * 128 + (c - 128)]; v = he * mbw + he * mfw; }
        else if (c < 384) v = Ws[aa[gj] * 128 + (c - 256)] * mbw;
        else v = hV[(size_t)gj * 128 + (c - 384)] * mbw + hVenc[(size_t)gj * 128 + (c - 384)] * mfw;
        A0[idx] = (_Float16)v;
    }
    __syncthreads();
    block_gemm<true, true, false, 2>(A0, 512, W1, 512, 128, b1, A1, nullptr, 128);
    __syncthreads();
    block_gemm<true, true, false, 2>(A1, 128, W2, 128, 128, b2, A0, nullptr, 128);
    __syncthreads();
    block_gemm<false, false, true, 2>(A0, 128, W3, 128, 128, b3, nullptr, Cm, 128);
    __syncthreads();
    if (tid < 128) {
        float s = 0.f;
        for (int e = 0; e < 32; ++e) s += Cm[e * 128 + tid];
        msum[(size_t)bi * 128 + tid] = s;
    }
}

// ---------------- node update (shared enc/dec) ----------------
__global__ void k_node_upd(float* __restrict__ hV, const float* __restrict__ msum,
                           const float* __restrict__ mask,
                           const float* n1g, const float* n1b,
                           const _Float16* Win, const float* bin,
                           const _Float16* Wout, const float* bout,
                           const float* n2g, const float* n2b) {
    __shared__ float    xln[16 * 128];
    __shared__ _Float16 a16[16 * 128];
    __shared__ _Float16 hid[16 * 512];
    __shared__ float    dh[16 * 128];
    int n0 = blockIdx.x * 16;
    int lane = threadIdx.x & 31, wv = threadIdx.x >> 5;
    for (int r2 = 0; r2 < 2; ++r2) {
        int r = wv * 2 + r2, node = n0 + r;
        float v[4], o[4];
#pragma unroll
        for (int t = 0; t < 4; ++t) {
            int c = lane + 32 * t;
            v[t] = hV[(size_t)node * 128 + c] + msum[(size_t)node * 128 + c] * (1.0f / 30.0f);
        }
        ln4(v, lane, n1g, n1b, o);
#pragma unroll
        for (int t = 0; t < 4; ++t) {
            int c = lane + 32 * t;
            xln[r * 128 + c] = o[t];
            a16[r * 128 + c] = (_Float16)o[t];
        }
    }
    __syncthreads();
    block_gemm<true, true, false, 1>(a16, 128, Win, 128, 512, bin, hid, nullptr, 512);
    __syncthreads();
    block_gemm<false, false, true, 1>(hid, 512, Wout, 512, 128, bout, nullptr, dh, 128);
    __syncthreads();
    for (int r2 = 0; r2 < 2; ++r2) {
        int r = wv * 2 + r2, node = n0 + r;
        float mk = mask[node];
        float v[4], o[4];
#pragma unroll
        for (int t = 0; t < 4; ++t) { int c = lane + 32 * t; v[t] = xln[r * 128 + c] + dh[r * 128 + c]; }
        ln4(v, lane, n2g, n2b, o);
#pragma unroll
        for (int t = 0; t < 4; ++t) hV[(size_t)node * 128 + lane + 32 * t] = o[t] * mk;
    }
}

// ---------------- head + pooling ----------------
__global__ void k_head(const float* __restrict__ hV, const float* __restrict__ mask,
                       const float* W1, const float* b1, const float* W2, const float* b2,
                       const float* W3, const float* b3, float* __restrict__ pr) {
    __shared__ float vbuf[128], h1[128], h2[128], red[128];
    int node = blockIdx.x, t = threadIdx.x;
    vbuf[t] = hV[(size_t)node * 128 + t];
    __syncthreads();
    float s = 0.f;
    for (int k = 0; k < 128; ++k) s += vbuf[k] * W1[k * 128 + t];
    h1[t] = gelu_f(s + b1[t]);
    __syncthreads();
    s = 0.f;
    for (int k = 0; k < 128; ++k) s += h1[k] * W2[k * 128 + t];
    h2[t] = gelu_f(s + b2[t]);
    __syncthreads();
    red[t] = h2[t] * W3[t];
    __syncthreads();
    for (int st = 64; st > 0; st >>= 1) {
        if (t < st) red[t] += red[t + st];
        __syncthreads();
    }
    if (t == 0) pr[node] = (red[0] + b3[0]) * mask[node];
}

__global__ void k_final(const float* __restrict__ pr, const float* __restrict__ mask, float* __restrict__ out) {
    __shared__ float s1[256], s2[256];
    int b = blockIdx.x, t = threadIdx.x;
    float a = 0.f, m = 0.f;
    for (int j = t; j < Lc; j += 256) { a += pr[b * Lc + j]; m += mask[b * Lc + j]; }
    s1[t] = a; s2[t] = m;
    __syncthreads();
    for (int st = 128; st > 0; st >>= 1) {
        if (t < st) { s1[t] += s1[t + st]; s2[t] += s2[t + st]; }
        __syncthreads();
    }
    if (t == 0) { float valid = s2[0] < 1.f ? 1.f : s2[0]; out[b] = s1[0] / sqrtf(valid); }
}

// ---------------- host ----------------
extern "C" void kernel_launch(void* const* d_in, const int* in_sizes, int n_in,
                              void* d_out, int out_size, void* d_ws, size_t ws_size,
                              hipStream_t stream) {
    (void)in_sizes; (void)n_in; (void)out_size; (void)ws_size;
    auto F  = [&](int i) { return (const float*)d_in[i]; };
    auto IP = [&](int i) { return (const int*)d_in[i]; };

    // input indices (setup_inputs dict order, params nested insertion order)
    const float* X = F(0);          const int* aa = IP(1);
    const float* mask = F(2);       const float* chainM = F(3);
    const int* ridx = IP(4);        const int* cenc = IP(5);
    const float* posW = F(6);       const float* posB = F(7);
    // We_feat=8, ne_g=9, ne_b=10, W_e=11, W_e_b=12, W_s=13
    // enc: 14..35 ; dec: 36..49 ; head: 50..55

    // workspace carve-up
    char* w = (char*)d_ws;
    size_t off = 0;
    auto alloc = [&](size_t bytes) { off = (off + 255) & ~(size_t)255; void* p = w + off; off += bytes; return p; };

    const size_t N16 = 1642496;   // total f16 weight elements
    _Float16* h16 = (_Float16*)alloc(N16 * sizeof(_Float16));
    int*   Eidx  = (int*)alloc((size_t)NEDGE * 4);
    float* Cb    = (float*)alloc((size_t)NNODE * 3 * 4);
    float* hV    = (float*)alloc((size_t)NNODE * Hc * 4);
    float* hVenc = (float*)alloc((size_t)NNODE * Hc * 4);
    float* hE    = (float*)alloc((size_t)NEDGE * Hc * 4);
    float* msum  = (float*)alloc((size_t)NNODE * Hc * 4);
    int*   rnk   = (int*)alloc((size_t)NNODE * 4);
    float* pr    = (float*)alloc((size_t)NNODE * 4);

    size_t o = 0;
    auto nx = [&](size_t n) { _Float16* p = h16 + o; o += n; return p; };
    _Float16* hWef  = nx(416 * 128);
    _Float16* hWe   = nx(128 * 128);
    _Float16* eW1   = nx((size_t)3 * 384 * 128);
    _Float16* eW2   = nx((size_t)3 * 128 * 128);
    _Float16* eW3   = nx((size_t)3 * 128 * 128);
    _Float16* eWin  = nx((size_t)3 * 128 * 512);
    _Float16* eWout = nx((size_t)3 * 512 * 128);
    _Float16* eW11  = nx((size_t)3 * 384 * 128);
    _Float16* eW12  = nx((size_t)3 * 128 * 128);
    _Float16* eW13  = nx((size_t)3 * 128 * 128);
    _Float16* dW1   = nx((size_t)3 * 512 * 128);
    _Float16* dW2   = nx((size_t)3 * 128 * 128);
    _Float16* dW3   = nx((size_t)3 * 128 * 128);
    _Float16* dWin  = nx((size_t)3 * 128 * 512);
    _Float16* dWout = nx((size_t)3 * 512 * 128);

    struct CvtJob { const float* src; _Float16* dst; int n; };
    CvtJob jobs[] = {
        {F(8),  hWef,  416 * 128}, {F(11), hWe, 128 * 128},
        {F(14), eW1,  3 * 384 * 128}, {F(16), eW2, 3 * 128 * 128}, {F(18), eW3, 3 * 128 * 128},
        {F(22), eWin, 3 * 128 * 512}, {F(24), eWout, 3 * 512 * 128},
        {F(28), eW11, 3 * 384 * 128}, {F(30), eW12, 3 * 128 * 128}, {F(32), eW13, 3 * 128 * 128},
        {F(36), dW1,  3 * 512 * 128}, {F(38), dW2, 3 * 128 * 128}, {F(40), dW3, 3 * 128 * 128},
        {F(44), dWin, 3 * 128 * 512}, {F(46), dWout, 3 * 512 * 128},
    };
    for (auto& j : jobs)
        k_cvt<<<(j.n + 255) / 256, 256, 0, stream>>>(j.src, j.dst, j.n);

    k_backbone<<<(NNODE + 255) / 256, 256, 0, stream>>>(X, Cb);
    k_topk<<<NNODE, 256, 0, stream>>>(X, mask, Eidx);
    k_edgefeat<<<NNODE, 256, 0, stream>>>(X, Cb, Eidx, ridx, cenc, posW, posB,
                                          hWef, F(9), F(10), hWe, F(12), hE);
    k_rank<<<(NNODE + 255) / 256, 256, 0, stream>>>(chainM, mask, rnk);
    hipMemsetAsync(hV, 0, (size_t)NNODE * Hc * 4, stream);

    for (int l = 0; l < 3; ++l) {
        k_enc_msg<<<NNODE, 256, 0, stream>>>(hV, hE, Eidx, mask,
            eW1 + (size_t)l * 384 * 128, F(15) + l * 128,
            eW2 + (size_t)l * 128 * 128, F(17) + l * 128,
            eW3 + (size_t)l * 128 * 128, F(19) + l * 128, msum);
        k_node_upd<<<NNODE / 16, 256, 0, stream>>>(hV, msum, mask,
            F(20) + l * 128, F(21) + l * 128,
            eWin + (size_t)l * 128 * 512, F(23) + l * 512,
            eWout + (size_t)l * 512 * 128, F(25) + l * 128,
            F(26) + l * 128, F(27) + l * 128);
        k_enc_edge<<<NNODE, 256, 0, stream>>>(hV, hE, Eidx,
            eW11 + (size_t)l * 384 * 128, F(29) + l * 128,
            eW12 + (size_t)l * 128 * 128, F(31) + l * 128,
            eW13 + (size_t)l * 128 * 128, F(33) + l * 128,
            F(34) + l * 128, F(35) + l * 128);
    }

    hipMemcpyAsync(hVenc, hV, (size_t)NNODE * Hc * 4, hipMemcpyDeviceToDevice, stream);

    for (int l = 0; l < 3; ++l) {
        k_dec_msg<<<NNODE, 256, 0, stream>>>(hV, hVenc, hE, Eidx, rnk, aa, mask, F(13),
            dW1 + (size_t)l * 512 * 128, F(37) + l * 128,
            dW2 + (size_t)l * 128 * 128, F(39) + l * 128,
            dW3 + (size_t)l * 128 * 128, F(41) + l * 128, msum);
        k_node_upd<<<NNODE / 16, 256, 0, stream>>>(hV, msum, mask,
            F(42) + l * 128, F(43) + l * 128,
            dWin + (size_t)l * 128 * 512, F(45) + l * 512,
            dWout + (size_t)l * 512 * 128, F(47) + l * 128,
            F(48) + l * 128, F(49) + l * 128);
    }

    k_head<<<NNODE, 128, 0, stream>>>(hV, mask, F(50), F(51), F(52), F(53), F(54), F(55), pr);
    k_final<<<Bc, 256, 0, stream>>>(pr, mask, (float*)d_out);
}
